// CirculantLinear_v6_11785390260451
// MI455X (gfx1250) — compile-verified, matrix operand order checked by reading, not run on hardware
//
#include <hip/hip_runtime.h>
#include <stdint.h>

typedef __attribute__((ext_vector_type(16))) _Float16 v16h;
typedef __attribute__((ext_vector_type(8)))  _Float16 v8h;
typedef __attribute__((ext_vector_type(8)))  float    v8f;

#define Bsz   8192
#define INC   1024
#define OUTC  1024
#define GXc   128
#define Mtot  (2 * Bsz)          // 16384 : re-plane then im-plane stacked on M
#define KST   64                 // K-halves per pipeline stage (2 WMMA k-steps)
#define LDW   72                 // LDS row stride in halves (64 + 8 pad)
#define TILEH (128 * LDW)        // halves per staged matrix tile
#define NSTG  (INC / KST)        // 16 pipeline stages

// ---------------------------------------------------------------------------
// Kernel 1: expand eigens into dense f16 weight matrix
//   W[o][i] = eigens[y, x, (n - j) & 7],  o = y*8+n, i = x*8+j  (circulant)
// ---------------------------------------------------------------------------
__global__ __launch_bounds__(256) void build_w_kernel(
    const float* __restrict__ eigens, _Float16* __restrict__ W16)
{
    int tid = blockIdx.x * 256 + threadIdx.x;       // 1M threads
    int o = tid >> 10;
    int i = tid & 1023;
    int y = o >> 3, n = o & 7;
    int x = i >> 3, j = i & 7;
    int k = (n - j) & 7;
    W16[tid] = (_Float16)eigens[((y * GXc + x) << 3) + k];
}

// ---------------------------------------------------------------------------
// Kernel 2: deinterleave x [B, IN, 2] f32  ->  Xh [2 planes][B][IN] f16
// ---------------------------------------------------------------------------
__global__ __launch_bounds__(256) void pack_x_kernel(
    const float2* __restrict__ x, _Float16* __restrict__ Xh)
{
    int tid = blockIdx.x * 256 + threadIdx.x;       // 8M threads: (b, i)
    float2 v = x[tid];
    Xh[tid]                     = (_Float16)v.x;    // re plane
    Xh[(size_t)Bsz * INC + tid] = (_Float16)v.y;    // im plane
}

// ---------------------------------------------------------------------------
// CDNA5 async DMA: global -> LDS, 16 bytes per lane, tracked by ASYNCcnt
// ---------------------------------------------------------------------------
__device__ __forceinline__ void async_ld_b128(uint32_t lds_off, const _Float16* g)
{
    asm volatile("global_load_async_to_lds_b128 %0, %1, off"
                 :: "v"(lds_off), "v"((uint64_t)(uintptr_t)g)
                 : "memory");
}

// ---------------------------------------------------------------------------
// Kernel 3: C[16384, 1024] = Xh * W^T via v_wmma_f32_16x16x32_f16
//   128x128 block tile, 64-wide K stage, double-buffered LDS fed by
//   global_load_async_to_lds_b128 (2-stage software pipeline).
// ---------------------------------------------------------------------------
__global__ __launch_bounds__(256) void gemm_kernel(
    const _Float16* __restrict__ Xh,
    const _Float16* __restrict__ W16,
    float* __restrict__ out)
{
    // [stage][A|B] tiles: 2 * 2 * 128*72 halves = 72 KB
    __shared__ __align__(16) _Float16 smem[2 * 2 * TILEH];

    const int nblock = blockIdx.x;      // 0..7   (N / 128)
    const int mblock = blockIdx.y;      // 0..127 (M / 128)

    const int t     = threadIdx.x;
    const int lane  = t & 31;
    const int wv    = t >> 5;           // 0..7
    const int waveM = wv & 3;           // 4 waves along M  -> 32 rows each
    const int waveN = wv >> 2;          // 2 waves along N  -> 64 cols each
    const int lrow  = lane & 15;        // fragment row/col within a 16-tile
    const int khi   = (lane >> 4) * 8;  // per-lane K base (CDNA5 16-bit layout)

    // cooperative DMA mapping: 2 threads per row, 32 halves (64 B = 4 b128) each
    const int ldRow  = t >> 1;
    const int ldPart = (t & 1) * 32;

    const _Float16* Ag = Xh  + (size_t)(mblock * 128 + ldRow) * INC + ldPart;
    const _Float16* Bg = W16 + (size_t)(nblock * 128 + ldRow) * INC + ldPart;

    // LDS byte addresses for this thread's DMA destinations (flat addr[31:0])
    const uint32_t sbase = (uint32_t)(uintptr_t)smem;
    const uint32_t rowoff = (uint32_t)(ldRow * LDW + ldPart) * 2u;
    uint32_t dstA[2], dstB[2];
#pragma unroll
    for (int s = 0; s < 2; ++s) {
        dstA[s] = sbase + (uint32_t)(s * 2 * TILEH) * 2u + rowoff;
        dstB[s] = sbase + (uint32_t)((s * 2 + 1) * TILEH) * 2u + rowoff;
    }

    v8f acc[2][4] = {};

    // ---- prologue: DMA stage 0 ----
#pragma unroll
    for (int c = 0; c < 4; ++c) {
        async_ld_b128(dstA[0] + c * 16, Ag + c * 8);
        async_ld_b128(dstB[0] + c * 16, Bg + c * 8);
    }

    for (int it = 0; it < NSTG; ++it) {
        const int cur = it & 1;

        if (it + 1 < NSTG) {
            // issue next stage into the other buffer (read-done via last barrier)
            const int knext = (it + 1) * KST;
#pragma unroll
            for (int c = 0; c < 4; ++c) {
                async_ld_b128(dstA[cur ^ 1] + c * 16, Ag + knext + c * 8);
                async_ld_b128(dstB[cur ^ 1] + c * 16, Bg + knext + c * 8);
            }
            asm volatile("s_wait_asynccnt 0x8" ::: "memory"); // stage `it` landed
        } else {
            asm volatile("s_wait_asynccnt 0x0" ::: "memory");
        }
        __syncthreads();

        const _Float16* As = smem + cur * 2 * TILEH;
        const _Float16* Bs = As + TILEH;

#pragma unroll
        for (int kk = 0; kk < 2; ++kk) {            // two WMMA k-steps per stage
            const int koff = kk * 32 + khi;
            v16h afr[2], bfr[4];
#pragma unroll
            for (int mt = 0; mt < 2; ++mt) {
                const _Float16* p = As + (waveM * 32 + mt * 16 + lrow) * LDW + koff;
                *(v8h*)&afr[mt]       = *(const v8h*)p;
                *((v8h*)&afr[mt] + 1) = *(const v8h*)(p + 16);
            }
#pragma unroll
            for (int nt = 0; nt < 4; ++nt) {
                const _Float16* p = Bs + (waveN * 64 + nt * 16 + lrow) * LDW + koff;
                *(v8h*)&bfr[nt]       = *(const v8h*)p;
                *((v8h*)&bfr[nt] + 1) = *(const v8h*)(p + 16);
            }
#pragma unroll
            for (int mt = 0; mt < 2; ++mt)
#pragma unroll
                for (int nt = 0; nt < 4; ++nt)
                    acc[mt][nt] = __builtin_amdgcn_wmma_f32_16x16x32_f16(
                        false, afr[mt], false, bfr[nt],
                        (short)0, acc[mt][nt], false, false);
        }
        __syncthreads();   // all reads of buf[cur] done before it is re-filled
    }

    // ---- epilogue: scatter into interleaved [B, OUT, 2] f32 output ----
    const int mrow = (lane >> 4) * 8;   // C/D layout: VGPR v -> M = v + 8*(lane/16)
    const int ncol = lane & 15;
#pragma unroll
    for (int mt = 0; mt < 2; ++mt) {
#pragma unroll
        for (int nt = 0; nt < 4; ++nt) {
            const int og = nblock * 128 + waveN * 64 + nt * 16 + ncol;
#pragma unroll
            for (int v = 0; v < 8; ++v) {
                const int mg = mblock * 128 + waveM * 32 + mt * 16 + mrow + v;
                const int c  = mg >> 13;        // 0 = re plane, 1 = im plane
                const int b  = mg & (Bsz - 1);
                out[(((size_t)b * OUTC + og) << 1) + c] = acc[mt][nt][v];
            }
        }
    }
}

// ---------------------------------------------------------------------------
extern "C" void kernel_launch(void* const* d_in, const int* in_sizes, int n_in,
                              void* d_out, int out_size, void* d_ws, size_t ws_size,
                              hipStream_t stream) {
    const float*  x      = (const float*)d_in[0];   // [8192, 1024, 2] f32
    const float*  eigens = (const float*)d_in[1];   // [128, 128, 8]   f32
    float*        out    = (float*)d_out;           // [8192, 1024, 2] f32

    _Float16* W16 = (_Float16*)d_ws;                                     // 2 MB
    _Float16* Xh  = (_Float16*)((char*)d_ws + ((size_t)OUTC * INC * 2)); // 32 MB

    build_w_kernel<<<(OUTC * INC) / 256, 256, 0, stream>>>(eigens, W16);
    pack_x_kernel<<<(Bsz * INC) / 256, 256, 0, stream>>>((const float2*)x, Xh);

    dim3 grid(OUTC / 128, Mtot / 128, 1);
    gemm_kernel<<<grid, 256, 0, stream>>>(Xh, W16, out);
}
